// ContrastiveLossELI5_57578331570319
// MI455X (gfx1250) — compile-verified, matrix operand order checked by reading, not run on hardware
//
#include <hip/hip_runtime.h>
#include <hip/hip_bf16.h>

#define N_ROWS   4096
#define DIM      512
#define TOT      8192                    // 2N
#define TEMP_INV 2.0f                    // 1/temperature
#define EXP_SCALE 2.8853900817779268f    // log2(e)/T
#define LN2       0.69314718055994530942f
#define SELF_TERM 7.38905609893065f      // exp(1/T) (diagonal of exp_sim)

typedef __attribute__((ext_vector_type(16))) _Float16 v16h;
typedef __attribute__((ext_vector_type(8)))  _Float16 v8h;
typedef __attribute__((ext_vector_type(4)))  _Float16 v4h;
typedef __attribute__((ext_vector_type(8)))  float    v8f;

union Frag16 { v16h v; v8h h[2]; };

// ---------------------------------------------------------------- zero init
__global__ void k_zero(float* __restrict__ rowsum, float* __restrict__ out) {
    int i = blockIdx.x * blockDim.x + threadIdx.x;
    if (i < TOT) rowsum[i] = 0.0f;
    if (i == 0)  out[0] = 0.0f;
}

// ----------------------------------------------- row-normalize f32 -> f16 z
__global__ void k_normalize(const float* __restrict__ emb_i,
                            const float* __restrict__ emb_j,
                            _Float16* __restrict__ z16) {
    const int row  = blockIdx.x * 8 + (threadIdx.x >> 5);
    const int lane = threadIdx.x & 31;
    const float* src = (row < N_ROWS) ? (emb_i + (size_t)row * DIM)
                                      : (emb_j + (size_t)(row - N_ROWS) * DIM);
    float4 vals[4];
    float ss = 0.0f;
    #pragma unroll
    for (int s = 0; s < 4; ++s) {
        vals[s] = ((const float4*)src)[lane + 32 * s];
        ss += vals[s].x * vals[s].x + vals[s].y * vals[s].y
            + vals[s].z * vals[s].z + vals[s].w * vals[s].w;
    }
    #pragma unroll
    for (int m = 1; m < 32; m <<= 1) ss += __shfl_xor(ss, m, 32);
    const float rn = __builtin_amdgcn_rsqf(ss);   // v_rsq_f32
    _Float16* dst = z16 + (size_t)row * DIM;
    #pragma unroll
    for (int s = 0; s < 4; ++s) {
        v4h h;
        h.x = (_Float16)(vals[s].x * rn);
        h.y = (_Float16)(vals[s].y * rn);
        h.z = (_Float16)(vals[s].z * rn);
        h.w = (_Float16)(vals[s].w * rn);
        ((v4h*)dst)[lane + 32 * s] = h;
    }
}

// ------------------------------------------- positive-pair dots (numerators)
__global__ void k_posdot(const _Float16* __restrict__ z16,
                         float* __restrict__ posdot) {
    const int p    = blockIdx.x * 8 + (threadIdx.x >> 5);
    const int lane = threadIdx.x & 31;
    const _Float16* a = z16 + (size_t)p * DIM + lane * 16;
    const _Float16* b = z16 + (size_t)(p + N_ROWS) * DIM + lane * 16;
    float s = 0.0f;
    #pragma unroll
    for (int e = 0; e < 16; ++e) s += (float)a[e] * (float)b[e];
    #pragma unroll
    for (int m = 1; m < 32; m <<= 1) s += __shfl_xor(s, m, 32);
    if (lane == 0) posdot[p] = s;
}

// --------------------------- main kernel: rowsum_r = sum_c exp(z_r.z_c / T)
// block = 8 waves; 128 rows/block (16/wave, A resident in 128 VGPRs).
// B macro-tile = 32 cols x 512 halves, LDS ping-pong (2 buffers, 1 barrier/it).
// Per iteration: issue next tile's global loads into staging regs, run the
// 32-WMMA loop (2 chains, distance-1 software-pipelined B frags), then
// ds_store the staged tile into the other buffer.
#define BS_STRIDE  520            // 512 + 8 halves pad (bank-conflict-free)
#define BUF_HALVES (32 * BS_STRIDE)

__global__ void __launch_bounds__(256)
k_simexp(const _Float16* __restrict__ z16, float* __restrict__ rowsum) {
    __shared__ _Float16 Bs[2 * BUF_HALVES];

    const int tid     = threadIdx.x;
    const int lane    = tid & 31;
    const int wv      = tid >> 5;
    const int rowbase = blockIdx.x * 128;
    const int ct0     = blockIdx.y * 32;   // 32-column macro-tiles
    const int ct1     = ct0 + 32;

    // WMMA 16-bit fragment addressing (wave32): lane&15 = matrix row/col,
    // half-wave selects which 8-half K sub-chunk the lane holds.
    const int m    = lane & 15;
    const int koff = (lane >> 4) * 8;

    // Preload A: this wave's 16 rows, full K=512 -> 16 fragments (128 VGPRs)
    Frag16 a[16];
    {
        const _Float16* arow = z16 + (size_t)(rowbase + wv * 16 + m) * DIM;
        #pragma unroll
        for (int kk = 0; kk < 16; ++kk) {
            a[kk].h[0] = *(const v8h*)(arow + kk * 32 + koff);
            a[kk].h[1] = *(const v8h*)(arow + kk * 32 + koff + 16);
        }
    }

    float acc[8];
    #pragma unroll
    for (int v = 0; v < 8; ++v) acc[v] = 0.0f;

    // cooperative copy mapping: 8 threads per column, 64 halves (128B) each
    const int cc  = tid >> 3;   // column 0..31
    const int seg = tid & 7;    // 64-half segment
    const int dstoff = cc * BS_STRIDE + seg * 64;

    v8h stage[8];
    // prologue: stage + store tile ct0 into buffer 0
    {
        const v8h* src = (const v8h*)(z16 + (size_t)(ct0 * 32 + cc) * DIM + seg * 64);
        #pragma unroll
        for (int q = 0; q < 8; ++q) stage[q] = src[q];
        v8h* d = (v8h*)(Bs + dstoff);
        #pragma unroll
        for (int q = 0; q < 8; ++q) d[q] = stage[q];
    }
    __syncthreads();

    for (int ct = ct0; ct < ct1; ++ct) {
        // 1) issue next tile's global loads first (hidden behind WMMA loop)
        if (ct + 1 < ct1) {
            const v8h* src = (const v8h*)(z16 + (size_t)((ct + 1) * 32 + cc) * DIM + seg * 64);
            #pragma unroll
            for (int q = 0; q < 8; ++q) stage[q] = src[q];
        }

        // 2) compute on current buffer: 2 chains, distance-1 pipelined B frags
        const _Float16* cur   = Bs + (ct & 1) * BUF_HALVES;
        const _Float16* b0row = cur + m * BS_STRIDE + koff;
        const _Float16* b1row = b0row + 16 * BS_STRIDE;

        v8f c0 = {}, c1 = {};
        Frag16 p0, p1;
        p0.h[0] = *(const v8h*)(b0row);
        p0.h[1] = *(const v8h*)(b0row + 16);
        p1.h[0] = *(const v8h*)(b1row);
        p1.h[1] = *(const v8h*)(b1row + 16);
        #pragma unroll
        for (int kk = 0; kk < 16; ++kk) {
            Frag16 n0, n1;
            if (kk < 15) {
                n0.h[0] = *(const v8h*)(b0row + (kk + 1) * 32);
                n0.h[1] = *(const v8h*)(b0row + (kk + 1) * 32 + 16);
                n1.h[0] = *(const v8h*)(b1row + (kk + 1) * 32);
                n1.h[1] = *(const v8h*)(b1row + (kk + 1) * 32 + 16);
            }
            c0 = __builtin_amdgcn_wmma_f32_16x16x32_f16(
                    false, a[kk].v, false, p0.v, (short)0, c0, false, false);
            c1 = __builtin_amdgcn_wmma_f32_16x16x32_f16(
                    false, a[kk].v, false, p1.v, (short)0, c1, false, false);
            p0 = n0;
            p1 = n1;
        }

        // fused epilogue: exp(sim/T) accumulated per (row-half, lane=column)
        #pragma unroll
        for (int v = 0; v < 8; ++v)
            acc[v] += __builtin_amdgcn_exp2f(c0[v] * EXP_SCALE)
                    + __builtin_amdgcn_exp2f(c1[v] * EXP_SCALE);

        // 3) spill staged tile into the other LDS buffer; single barrier/iter
        if (ct + 1 < ct1) {
            v8h* d = (v8h*)(Bs + ((ct + 1) & 1) * BUF_HALVES + dstoff);
            #pragma unroll
            for (int q = 0; q < 8; ++q) d[q] = stage[q];
        }
        __syncthreads();
    }

    // reduce 16 columns (lanes) per half-wave -> one partial rowsum per row
    #pragma unroll
    for (int v = 0; v < 8; ++v) {
        float s = acc[v];
        s += __shfl_xor(s, 1, 32);
        s += __shfl_xor(s, 2, 32);
        s += __shfl_xor(s, 4, 32);
        s += __shfl_xor(s, 8, 32);
        if ((lane & 15) == 0) {
            const int row = rowbase + wv * 16 + v + (lane >> 4) * 8;
            unsafeAtomicAdd(&rowsum[row], s);   // global_atomic_add_f32
        }
    }
}

// ------------------------------------------------------------- final scalar
__global__ void k_finalize(const float* __restrict__ rowsum,
                           const float* __restrict__ posdot,
                           float* __restrict__ out) {
    __shared__ float red[256];
    const int i = blockIdx.x * 256 + threadIdx.x;
    const float denom = rowsum[i] - SELF_TERM;
    const float pos   = (i < N_ROWS) ? posdot[i] : posdot[i - N_ROWS];
    const float loss  = __builtin_amdgcn_logf(denom) * LN2 - pos * TEMP_INV;
    red[threadIdx.x] = loss;
    __syncthreads();
    for (int s = 128; s > 0; s >>= 1) {
        if (threadIdx.x < s) red[threadIdx.x] += red[threadIdx.x + s];
        __syncthreads();
    }
    if (threadIdx.x == 0) unsafeAtomicAdd(out, red[0] * (1.0f / TOT));
}

// ----------------------------------------------------------------- launcher
extern "C" void kernel_launch(void* const* d_in, const int* in_sizes, int n_in,
                              void* d_out, int out_size, void* d_ws, size_t ws_size,
                              hipStream_t stream) {
    (void)in_sizes; (void)n_in; (void)out_size; (void)ws_size;
    const float* emb_i = (const float*)d_in[0];
    const float* emb_j = (const float*)d_in[1];
    float* out = (float*)d_out;

    char* ws = (char*)d_ws;
    _Float16* z16    = (_Float16*)ws;                                // 8 MB
    float*    rowsum = (float*)(ws + (size_t)TOT * DIM * 2);         // 32 KB
    float*    posdot = (float*)(ws + (size_t)TOT * DIM * 2 + (size_t)TOT * 4);

    k_zero     <<<TOT / 256, 256, 0, stream>>>(rowsum, out);
    k_normalize<<<TOT / 8,   256, 0, stream>>>(emb_i, emb_j, z16);
    k_posdot   <<<N_ROWS / 8, 256, 0, stream>>>(z16, posdot);
    dim3 grid(TOT / 128, 8);   // 64 row-groups x 8 column-groups = 512 WGs
    k_simexp   <<<grid, 256, 0, stream>>>(z16, rowsum);
    k_finalize <<<TOT / 256, 256, 0, stream>>>(rowsum, posdot, out);
}
